// SmoothQuantSegmentLinear_30150670418081
// MI455X (gfx1250) — compile-verified, matrix operand order checked by reading, not run on hardware
//
#include <hip/hip_runtime.h>
#include <hip/hip_bf16.h>

typedef __attribute__((ext_vector_type(8))) int v8i;

#define CHUNKS 4
#define M_     8192      // B*S
#define K_     4096      // IN
#define OUT_   4096
#define OC_    1024      // OUT / CHUNKS
#define KT_    64        // K_/64 K-tiles
#define QX_DW_PER_CHUNK  (M_ * K_ / 4)       // dwords of packed int8 A per chunk
#define QW_DW_PER_CHUNK  (OC_ * K_ / 4)
#define QX_V8_PER_CHUNK  (M_ * K_ / 32)      // v8i (32B) fragments per chunk
#define QW_V8_PER_CHUNK  (OC_ * K_ / 32)
#define QX_BYTES   ((size_t)CHUNKS * M_ * K_)   // 128 MiB

// int8 symmetric quant, round-half-even like jnp.round, clip to [-127,127]
__device__ __forceinline__ unsigned q8(float v) {
    float r = rintf(v);
    r = fminf(127.0f, fmaxf(-127.0f, r));
    return (unsigned)((int)r) & 0xffu;
}

// WMMA 8-bit A/B swizzle: byte K within a 64-wide tile lives at
// lane = h*16 + row, dword v, byte b with K = (v/2)*16 + h*8 + (v%2)*4 + b.
// Four consecutive K (4-aligned) share one dword -> pack and store 32 bits.

__global__ __launch_bounds__(256) void quantX_kernel(
        const float* __restrict__ x, const float* __restrict__ smooth,
        const float* __restrict__ in_scales, int* __restrict__ qx) {
    int idx = blockIdx.x * 256 + threadIdx.x;   // one per (m, k/4)
    int m  = idx >> 10;                         // K_/4 == 1024
    int k  = (idx & 1023) << 2;
    float4 xv = *(const float4*)(x + (size_t)m * K_ + k);
    int tileM = m >> 4, r = m & 15;
    int kt = k >> 6;
    int q  = (k & 63) >> 2;
    int v  = ((q >> 2) << 1) | (q & 1);
    int h  = (q >> 1) & 1;
    int lane = (h << 4) | r;
    size_t didx = (size_t)(tileM * KT_ + kt) * 256 + lane * 8 + v;   // dword index
#pragma unroll
    for (int c = 0; c < CHUNKS; ++c) {
        float4 sv = *(const float4*)(smooth + c * K_ + k);
        float inv = __builtin_amdgcn_rcpf(in_scales[c]);
        unsigned p =  q8(xv.x * __builtin_amdgcn_rcpf(sv.x) * inv)
                   | (q8(xv.y * __builtin_amdgcn_rcpf(sv.y) * inv) << 8)
                   | (q8(xv.z * __builtin_amdgcn_rcpf(sv.z) * inv) << 16)
                   | (q8(xv.w * __builtin_amdgcn_rcpf(sv.w) * inv) << 24);
        qx[(size_t)c * QX_DW_PER_CHUNK + didx] = (int)p;
    }
}

__global__ __launch_bounds__(256) void quantW_kernel(
        const float* __restrict__ w, const float* __restrict__ smooth,
        const float* __restrict__ w_scales, int* __restrict__ qw) {
    int idx = blockIdx.x * 256 + threadIdx.x;   // one per (o_global, k/4)
    int og = idx >> 10;
    int k  = (idx & 1023) << 2;
    int c  = og >> 10;                          // OC_ == 1024
    int oc = og & 1023;
    float4 wv = *(const float4*)(w + (size_t)og * K_ + k);
    float4 sv = *(const float4*)(smooth + c * K_ + k);
    float inv = __builtin_amdgcn_rcpf(w_scales[c]);
    unsigned p =  q8(wv.x * sv.x * inv)
               | (q8(wv.y * sv.y * inv) << 8)
               | (q8(wv.z * sv.z * inv) << 16)
               | (q8(wv.w * sv.w * inv) << 24);
    int tileN = oc >> 4, r = oc & 15;
    int kt = k >> 6;
    int q  = (k & 63) >> 2;
    int v  = ((q >> 2) << 1) | (q & 1);
    int h  = (q >> 1) & 1;
    int lane = (h << 4) | r;
    size_t didx = (size_t)(tileN * KT_ + kt) * 256 + lane * 8 + v;
    qw[(size_t)c * QW_DW_PER_CHUNK + didx] = (int)p;
}

// 128x128 output tile per 256-thread block; 8 waves in a 2(M) x 4(N) grid.
// Each wave: 64x32 via 4x2 v_wmma_i32_16x16x64_iu8 accumulators, K-loop of 64.
__global__ __launch_bounds__(256) void gemm_iu8_kernel(
        const v8i* __restrict__ qx, const v8i* __restrict__ qw,
        const float* __restrict__ in_scales, const float* __restrict__ w_scales,
        const float* __restrict__ bias, float* __restrict__ out) {
    const int c  = blockIdx.y >> 3;       // chunk
    const int nb = blockIdx.y & 7;        // N-block within chunk
    const int wave  = threadIdx.x >> 5;
    const int lane  = threadIdx.x & 31;
    const int waveM = wave & 1;
    const int waveN = wave >> 1;
    const int rowTile0 = blockIdx.x * 8 + waveM * 4;   // 16-row tiles (of 512)
    const int colTile0 = nb * 8 + waveN * 2;           // 16-col tiles within chunk (of 64)

    const v8i* Aq = qx + (size_t)c * QX_V8_PER_CHUNK + lane;
    const v8i* Bq = qw + (size_t)c * QW_V8_PER_CHUNK + lane;

    v8i zero = {0, 0, 0, 0, 0, 0, 0, 0};
    v8i acc[4][2];
#pragma unroll
    for (int mi = 0; mi < 4; ++mi)
#pragma unroll
        for (int ni = 0; ni < 2; ++ni) acc[mi][ni] = zero;

    for (int kt = 0; kt < KT_; ++kt) {
        v8i a[4], b[2];
#pragma unroll
        for (int mi = 0; mi < 4; ++mi)
            a[mi] = Aq[((size_t)(rowTile0 + mi) * KT_ + kt) * 32];
#pragma unroll
        for (int ni = 0; ni < 2; ++ni)
            b[ni] = Bq[((size_t)(colTile0 + ni) * KT_ + kt) * 32];
#pragma unroll
        for (int mi = 0; mi < 4; ++mi)
#pragma unroll
            for (int ni = 0; ni < 2; ++ni)
                acc[mi][ni] = __builtin_amdgcn_wmma_i32_16x16x64_iu8(
                    /*sgn_a=*/true, a[mi], /*sgn_b=*/true, b[ni],
                    acc[mi][ni], /*reuse_a=*/false, /*reuse_b=*/false);
    }

    const float scale = in_scales[c] * w_scales[c];
    const int h  = lane >> 4;
    const int nl = lane & 15;
#pragma unroll
    for (int mi = 0; mi < 4; ++mi) {
#pragma unroll
        for (int ni = 0; ni < 2; ++ni) {
            int col = c * OC_ + (colTile0 + ni) * 16 + nl;
            float bv = bias[col];
#pragma unroll
            for (int j = 0; j < 8; ++j) {
                int row = (rowTile0 + mi) * 16 + h * 8 + j;
                out[(size_t)row * OUT_ + col] = (float)acc[mi][ni][j] * scale + bv;
            }
        }
    }
}

extern "C" void kernel_launch(void* const* d_in, const int* in_sizes, int n_in,
                              void* d_out, int out_size, void* d_ws, size_t ws_size,
                              hipStream_t stream) {
    const float* x         = (const float*)d_in[0];
    const float* weight    = (const float*)d_in[1];
    const float* bias      = (const float*)d_in[2];
    const float* smooth    = (const float*)d_in[3];
    const float* in_scales = (const float*)d_in[4];
    const float* w_scales  = (const float*)d_in[5];

    int* qx = (int*)d_ws;                               // 128 MiB swizzled int8 A (x4 chunks)
    int* qw = (int*)((char*)d_ws + QX_BYTES);           // 16 MiB swizzled int8 B (x4 chunks)

    quantX_kernel<<<M_ * K_ / 4 / 256, 256, 0, stream>>>(x, smooth, in_scales, qx);
    quantW_kernel<<<OUT_ * K_ / 4 / 256, 256, 0, stream>>>(weight, smooth, w_scales, qw);

    dim3 grid(M_ / 128, (OUT_ / 128));                  // 64 x 32; blockIdx.y folds chunk
    gemm_iu8_kernel<<<grid, 256, 0, stream>>>((const v8i*)qx, (const v8i*)qw,
                                              in_scales, w_scales, bias, (float*)d_out);
}
